// MemoryEfficientSAM2Attention_1657857376483
// MI455X (gfx1250) — matrix operand order repaired and checked by reading, not compiled, not run on hardware
//
#include <hip/hip_runtime.h>
#include <hip/hip_bf16.h>

typedef __attribute__((ext_vector_type(16))) _Float16 v16h;
typedef __attribute__((ext_vector_type(8)))  float    v8f;

#define N_SEQ  8000
#define C_DIM  256
#define D_HEAD 32
#define BN     32                 // keys per flash block (one K-dim of the PV wmma)
#define NKB    (N_SEQ / BN)       // 250 key blocks
#define NQT    (N_SEQ / 16)       // 500 query tiles (16 rows each)
#define QSCALE 0.17677669529663687f   // 1/sqrt(32)

union FragH { v16h h; uint4 q[2]; };

// ---------------------------------------------------------------------------
// DPP16 cross-lane helpers: reduce across each 16-lane half of the wave with
// zero LDS traffic (replaces ds_bpermute + s_wait_dscnt serialization).
//   0xB1  = quad_perm [1,0,3,2]   (swap neighbors; xor 1)
//   0x4E  = quad_perm [2,3,0,1]   (swap pairs;     xor 2)
//   0x141 = row_half_mirror       (combine quads within 8)
//   0x140 = row_mirror            (combine 8-groups within 16)
// ---------------------------------------------------------------------------
template <int CTRL>
__device__ __forceinline__ float dpp_mov_f32(float v) {
    int i = __builtin_bit_cast(int, v);
    int r = __builtin_amdgcn_update_dpp(i, i, CTRL, 0xf, 0xf, true);
    return __builtin_bit_cast(float, r);
}
__device__ __forceinline__ float row16_max(float v) {
    v = fmaxf(v, dpp_mov_f32<0xB1>(v));
    v = fmaxf(v, dpp_mov_f32<0x4E>(v));
    v = fmaxf(v, dpp_mov_f32<0x141>(v));
    v = fmaxf(v, dpp_mov_f32<0x140>(v));
    return v;
}
__device__ __forceinline__ float row16_sum(float v) {
    v += dpp_mov_f32<0xB1>(v);
    v += dpp_mov_f32<0x4E>(v);
    v += dpp_mov_f32<0x141>(v);
    v += dpp_mov_f32<0x140>(v);
    return v;
}

// ---------------------------------------------------------------------------
// Stage 1: pointwise (1x1x1 conv) projections.
//   qf16 [N,32]  row-major f16, scale folded in
//   kf16 [N,32]  row-major f16
//   vfrag       f16 in WMMA B-matrix fragment layout per 32-key block:
//               flat = ((blk*16 + tile)*32 + lane)*16 + r*2 + elem
//               where s = 2r + elem, kk(feature-of-K-dim) = 8*(lane>=16) + (s&7) + 16*(s>>3),
//                     col = tile*16 + (lane&15)       (ISA 7.12.2 B layout)
// ---------------------------------------------------------------------------
__global__ void proj_kernel(const float* __restrict__ x,
                            const float* __restrict__ Wq, const float* __restrict__ bq,
                            const float* __restrict__ Wk, const float* __restrict__ bk,
                            const float* __restrict__ Wv, const float* __restrict__ bv,
                            _Float16* __restrict__ qf, _Float16* __restrict__ kf,
                            _Float16* __restrict__ vfrag)
{
    int idx = blockIdx.x * blockDim.x + threadIdx.x;
    if (idx >= N_SEQ * 320) return;
    int n = idx / 320;
    int e = idx % 320;               // 0..31 q, 32..63 k, 64..319 v
    const float* xr = x + (size_t)n * C_DIM;

    const float* w;
    float acc;
    int ld;
    if (e < 32)       { w = Wq + e;        acc = bq[e];      ld = 32;  }
    else if (e < 64)  { w = Wk + (e - 32); acc = bk[e - 32]; ld = 32;  }
    else              { w = Wv + (e - 64); acc = bv[e - 64]; ld = 256; }

#pragma unroll 8
    for (int c = 0; c < C_DIM; ++c)
        acc += xr[c] * w[c * ld];

    if (e < 32) {
        qf[(size_t)n * 32 + e] = (_Float16)(acc * QSCALE);
    } else if (e < 64) {
        kf[(size_t)n * 32 + (e - 32)] = (_Float16)acc;
    } else {
        int c    = e - 64;
        int blk  = n / 32,  kk = n % 32;
        int tile = c / 16,  ncol = c % 16;
        int half = (kk >> 3) & 1;
        int r    = ((kk & 7) | ((kk >> 4) << 3)) >> 1;
        int elem = kk & 1;
        int lane = half * 16 + ncol;
        size_t off = ((size_t)(blk * 16 + tile) * 32 + lane) * 16 + r * 2 + elem;
        vfrag[off] = (_Float16)acc;
    }
}

// ---------------------------------------------------------------------------
// Stage 2: flash attention. One wave per 16-query tile, 4 waves per block.
// ---------------------------------------------------------------------------
__global__ void __launch_bounds__(128) flash_kernel(
    const float* __restrict__ x, const float* __restrict__ gamma,
    const _Float16* __restrict__ qf, const _Float16* __restrict__ kf,
    const _Float16* __restrict__ vfrag, float* __restrict__ out)
{
    __shared__ __align__(16) _Float16 plds[4][16][32];   // per-wave P re-layout buffer

    const int lane = threadIdx.x & 31;
    const int wave = threadIdx.x >> 5;
    const int qt   = blockIdx.x * 4 + wave;
    if (qt >= NQT) return;                 // uniform per wave; EXEC stays all-1s
    const int q0   = qt * 16;
    const int half = lane >> 4;            // 0: lanes 0-15, 1: lanes 16-31
    const int l15  = lane & 15;

    // ---- Q as A-fragment: lane row = l15, cols [b,b+8) U [b+16,b+24), b=8*half
    FragH qa;
    {
        const uint4* p = (const uint4*)(qf + (size_t)(q0 + l15) * 32 + half * 8);
        qa.q[0] = p[0];
        qa.q[1] = p[2];                    // +32B = cols +16
    }

    v8f acc[16];
    float m[8], l[8];
#pragma unroll
    for (int t = 0; t < 16; ++t) { v8f z = {}; acc[t] = z; }
#pragma unroll
    for (int r = 0; r < 8; ++r) { m[r] = -1e30f; l[r] = 0.0f; }

    for (int j = 0; j < NKB; ++j) {
        // ---- K as two B-fragments: lane col = key j*32 + (tile*16 + l15),
        //      K-dim (feature) cols [b,b+8) U [b+16,b+24), b=8*half, as for Q, from row-major kf16.
        FragH kb0, kb1;
        {
            const uint4* p0 = (const uint4*)(kf + (size_t)(j * 32 + l15)      * 32 + half * 8);
            const uint4* p1 = (const uint4*)(kf + (size_t)(j * 32 + 16 + l15) * 32 + half * 8);
            kb0.q[0] = p0[0]; kb0.q[1] = p0[2];
            kb1.q[0] = p1[0]; kb1.q[1] = p1[2];
        }
        // Prefetch next key block's K rows (global_prefetch_b8) to hide latency
        // behind the 16-WMMA PV chain below.
        if (j + 1 < NKB) {
            __builtin_prefetch(kf + (size_t)((j + 1) * 32 + lane) * 32, 0, 0);
        }

        v8f zero = {};
        v8f s0 = __builtin_amdgcn_wmma_f32_16x16x32_f16(false, qa.h, false, kb0.h,
                                                        (short)0, zero, false, false);
        v8f s1 = __builtin_amdgcn_wmma_f32_16x16x32_f16(false, qa.h, false, kb1.h,
                                                        (short)0, zero, false, false);

        // ---- online softmax: rows live in VGPR index r (+8 for upper half),
        //      16 columns live across each 16-lane half -> DPP16 reductions.
        float p0a[8], p1a[8], alpha[8];
#pragma unroll
        for (int r = 0; r < 8; ++r) {
            float tm   = row16_max(fmaxf(s0[r], s1[r]));
            float mnew = fmaxf(m[r], tm);
            alpha[r]   = __expf(m[r] - mnew);
            float e0   = __expf(s0[r] - mnew);
            float e1   = __expf(s1[r] - mnew);
            float rs   = row16_sum(e0 + e1);
            l[r] = l[r] * alpha[r] + rs;
            m[r] = mnew;
            p0a[r] = e0; p1a[r] = e1;
        }
#pragma unroll
        for (int t = 0; t < 16; ++t)
#pragma unroll
            for (int r = 0; r < 8; ++r) acc[t][r] *= alpha[r];

        // ---- re-layout P (C-layout) -> A-fragment via per-wave LDS slice.
        //      Same-wave LDS ops are in-order; no barrier needed.
#pragma unroll
        for (int r = 0; r < 8; ++r) {
            int row = half * 8 + r;
            plds[wave][row][l15]      = (_Float16)p0a[r];
            plds[wave][row][l15 + 16] = (_Float16)p1a[r];
        }
        FragH pa;
        {
            const uint4* p = (const uint4*)&plds[wave][l15][half * 8];
            pa.q[0] = p[0];
            pa.q[1] = p[2];                // +32B = cols +16
        }

        // ---- O += P(16x32) @ V(32x256): 16 column tiles, V pre-swizzled so each
        //      lane's 16 f16 are contiguous (two global_load_b128).
#pragma unroll
        for (int t = 0; t < 16; ++t) {
            FragH vb;
            const uint4* pv = (const uint4*)vfrag
                            + ((size_t)(j * 16 + t) * 32 + lane) * 2;
            vb.q[0] = pv[0]; vb.q[1] = pv[1];
            acc[t] = __builtin_amdgcn_wmma_f32_16x16x32_f16(false, pa.h, false, vb.h,
                                                            (short)0, acc[t], false, false);
        }
    }

    // ---- epilogue: out = x + gamma * (acc / l)
    const float ga = gamma[0];
#pragma unroll
    for (int r = 0; r < 8; ++r) {
        int   row = q0 + half * 8 + r;
        float inv = 1.0f / l[r];
#pragma unroll
        for (int t = 0; t < 16; ++t) {
            size_t o = (size_t)row * C_DIM + t * 16 + l15;
            out[o] = x[o] + ga * (acc[t][r] * inv);
        }
    }
}

// ---------------------------------------------------------------------------
extern "C" void kernel_launch(void* const* d_in, const int* in_sizes, int n_in,
                              void* d_out, int out_size, void* d_ws, size_t ws_size,
                              hipStream_t stream)
{
    const float* x     = (const float*)d_in[0];
    const float* Wq    = (const float*)d_in[1];
    const float* bq    = (const float*)d_in[2];
    const float* Wk    = (const float*)d_in[3];
    const float* bk    = (const float*)d_in[4];
    const float* Wv    = (const float*)d_in[5];
    const float* bv    = (const float*)d_in[6];
    const float* gamma = (const float*)d_in[7];
    float* out = (float*)d_out;

    _Float16* qf    = (_Float16*)d_ws;                 // 8000*32  = 512 KB
    _Float16* kf    = qf + (size_t)N_SEQ * 32;         // 8000*32  = 512 KB
    _Float16* vfrag = kf + (size_t)N_SEQ * 32;         // 8000*256 = 4 MB

    int projN = N_SEQ * 320;
    proj_kernel<<<(projN + 255) / 256, 256, 0, stream>>>(
        x, Wq, bq, Wk, bk, Wv, bv, qf, kf, vfrag);

    flash_kernel<<<NQT / 4, 128, 0, stream>>>(x, gamma, qf, kf, vfrag, out);
}
